// CameraSelfAttention_46325517255100
// MI455X (gfx1250) — matrix operand order repaired and checked by reading, not compiled, not run on hardware
//
#include <hip/hip_runtime.h>

#define B_    8
#define CAMS_ 1024
#define E_    1024
#define H_    16
#define HD_   64
#define EPSV  1e-5f

typedef __attribute__((ext_vector_type(16))) __bf16 v16bf;
typedef __attribute__((ext_vector_type(8)))  float  v8f;

union AOp { v16bf v; unsigned int u[8]; };

__device__ inline unsigned int packbf2(float a, float b) {
  union { __bf16 h[2]; unsigned int u; } x;
  x.h[0] = (__bf16)a; x.h[1] = (__bf16)b;
  return x.u;
}
__device__ inline unsigned short f2bf(float f) {
  union { __bf16 h; unsigned short s; } x;
  x.h = (__bf16)f;
  return x.s;
}
__device__ inline v8f wmma_bf16(const AOp& a, const AOp& b, v8f c) {
  return __builtin_amdgcn_wmma_f32_16x16x32_bf16(
      false, a.v, false, b.v, (short)0, c, false, false);
}

// A operand (16x32 bf16): lane = M + 16*kblk; vgpr v holds K = 16*kblk + 2v, +1
// B operand (32x16 bf16): lane = N + 16*kblk; vgpr v holds K = 16*kblk + 2v, +1
// C/D (16x16 f32): vgpr r, lane l -> M = r + 8*(l>=16), N = l%16

// ---------------------------------------------------------------------------
// Kernel 0: fp32 -> bf16 conversion (vectorized, grid-stride free, exact fit)
// ---------------------------------------------------------------------------
__global__ __launch_bounds__(256) void cvt_kernel(
    const float* __restrict__ src, unsigned short* __restrict__ dst, int n4) {
  int i = blockIdx.x * 256 + threadIdx.x;
  if (i < n4) {
    float4 f = ((const float4*)src)[i];
    uint2 o;
    o.x = packbf2(f.x, f.y);
    o.y = packbf2(f.z, f.w);
    ((uint2*)dst)[i] = o;
  }
}

// ---------------------------------------------------------------------------
// Kernel 1: QKV GEMM (bf16 in, f32 acc) + SE3 projection + LayerNorm
// grid (CAMS/16, H, B), block 32
// ---------------------------------------------------------------------------
__global__ __launch_bounds__(32) void qkv_kernel(
    const unsigned short* __restrict__ Xb, const float* __restrict__ VM,
    const unsigned short* __restrict__ Wqb, const float* __restrict__ bq,
    const unsigned short* __restrict__ Wkb, const float* __restrict__ bk,
    const unsigned short* __restrict__ Wvb, const float* __restrict__ bv,
    unsigned short* __restrict__ qr,   // (B,H,C,64) bf16, layernormed q_rot
    unsigned short* __restrict__ kr,   // (B,H,C,64) bf16, layernormed k_rot
    unsigned short* __restrict__ vt)   // (B,H,64,C) bf16, v_rot transposed
{
  __shared__ float lds[3][16][64];
  const int lane = threadIdx.x;
  const int m  = lane & 15;
  const int kb = lane >> 4;
  const int c0 = blockIdx.x * 16;
  const int h  = blockIdx.y;
  const int b  = blockIdx.z;

  const v8f zero8 = {0.f,0.f,0.f,0.f,0.f,0.f,0.f,0.f};
  v8f acc[12];
#pragma unroll
  for (int i = 0; i < 12; ++i) acc[i] = zero8;

  const unsigned short* Ws[3] = {Wqb, Wkb, Wvb};
  const unsigned short* xrow = Xb + ((size_t)b * CAMS_ + c0 + m) * E_;

  for (int kc = 0; kc < E_; kc += 32) {
    AOp A;
#pragma unroll
    for (int v = 0; v < 8; ++v) {
      int k = 16 * kb + 2 * v;
      A.u[v] = *(const unsigned int*)(xrow + kc + k);
    }
#pragma unroll
    for (int s = 0; s < 3; ++s) {
#pragma unroll
      for (int t = 0; t < 4; ++t) {
        const unsigned short* wrow =
            Ws[s] + (size_t)(h * 64 + t * 16 + m) * E_ + kc + kb * 16;
        AOp Bm;
#pragma unroll
        for (int v = 0; v < 8; ++v)
          Bm.u[v] = *(const unsigned int*)(wrow + v * 2);
        acc[s * 4 + t] = wmma_bf16(A, Bm, acc[s * 4 + t]);
      }
    }
  }

  // spill accumulators (+ bias) to LDS as (stream, token, feature)
#pragma unroll
  for (int s = 0; s < 3; ++s) {
    const float* bias = (s == 0) ? bq : (s == 1) ? bk : bv;
#pragma unroll
    for (int t = 0; t < 4; ++t) {
      int feat = t * 16 + m;
      float bb = bias[h * 64 + feat];
#pragma unroll
      for (int r = 0; r < 8; ++r)
        lds[s][r + 8 * kb][feat] = acc[s * 4 + t][r] + bb;
    }
  }
  __syncthreads();

  // per-camera SE3 transform on groups of 4, then layernorm (q,k)
  const int tk = m;       // token handled by this lane
  const int half = kb;    // which 32-feature half
  const float* Pm = VM + ((size_t)b * CAMS_ + c0 + tk) * 16;
  float P[4][4];
#pragma unroll
  for (int i = 0; i < 4; ++i)
#pragma unroll
    for (int j = 0; j < 4; ++j) P[i][j] = Pm[i * 4 + j];

  float Ms[2][4][4];  // [0]=P^T (for q), [1]=P_inv (for k,v)
#pragma unroll
  for (int i = 0; i < 4; ++i)
#pragma unroll
    for (int j = 0; j < 4; ++j) Ms[0][i][j] = P[j][i];
#pragma unroll
  for (int i = 0; i < 3; ++i) {
#pragma unroll
    for (int j = 0; j < 3; ++j) Ms[1][i][j] = P[j][i];
    float ti = 0.f;
#pragma unroll
    for (int j = 0; j < 3; ++j) ti += P[j][i] * P[j][3];
    Ms[1][i][3] = -ti;
  }
  Ms[1][3][0] = 0.f; Ms[1][3][1] = 0.f; Ms[1][3][2] = 0.f; Ms[1][3][3] = 1.f;

  float sum[2] = {0.f, 0.f}, ssq[2] = {0.f, 0.f};
#pragma unroll
  for (int s = 0; s < 3; ++s) {
    const int mi = (s == 0) ? 0 : 1;
#pragma unroll
    for (int g = 0; g < 8; ++g) {
      int gi = half * 8 + g;
      float x0 = lds[s][tk][4 * gi + 0];
      float x1 = lds[s][tk][4 * gi + 1];
      float x2 = lds[s][tk][4 * gi + 2];
      float x3 = lds[s][tk][4 * gi + 3];
      float y[4];
#pragma unroll
      for (int i = 0; i < 4; ++i)
        y[i] = Ms[mi][i][0] * x0 + Ms[mi][i][1] * x1 +
               Ms[mi][i][2] * x2 + Ms[mi][i][3] * x3;
#pragma unroll
      for (int i = 0; i < 4; ++i) {
        lds[s][tk][4 * gi + i] = y[i];
        if (s < 2) { sum[s] += y[i]; ssq[s] += y[i] * y[i]; }
      }
    }
  }

#pragma unroll
  for (int s = 0; s < 2; ++s) {
    float tsum = sum[s] + __shfl_xor(sum[s], 16);
    float tssq = ssq[s] + __shfl_xor(ssq[s], 16);
    float mean = tsum * (1.f / 64.f);
    float var  = tssq * (1.f / 64.f) - mean * mean;
    float rstd = rsqrtf(var + EPSV);
    unsigned short* dst = (s == 0) ? qr : kr;
    size_t base = (((size_t)b * H_ + h) * CAMS_ + c0 + tk) * 64;
#pragma unroll
    for (int d0 = 0; d0 < 32; ++d0) {
      int d = half * 32 + d0;
      dst[base + d] = f2bf((lds[s][tk][d] - mean) * rstd);
    }
  }
  {
    size_t vbase = ((size_t)b * H_ + h) * 64;
#pragma unroll
    for (int d0 = 0; d0 < 32; ++d0) {
      int d = half * 32 + d0;
      vt[(vbase + d) * CAMS_ + c0 + tk] = f2bf(lds[2][tk][d]);
    }
  }
}

// ---------------------------------------------------------------------------
// Kernel 2: flash attention (16 queries / wave, 32 keys / step) + output proj
// grid (CAMS/16, H, B), block 32
// ---------------------------------------------------------------------------
__global__ __launch_bounds__(32) void attn_kernel(
    const unsigned short* __restrict__ qr, const unsigned short* __restrict__ kr,
    const unsigned short* __restrict__ vt, const float* __restrict__ VM,
    unsigned short* __restrict__ merged)   // (B,C,E) bf16
{
  __shared__ __align__(4) unsigned short p_lds[16][32];
  __shared__ float o_lds[16][64];
  const int lane = threadIdx.x;
  const int m  = lane & 15;
  const int kb = lane >> 4;
  const int c0 = blockIdx.x * 16;
  const int h  = blockIdx.y;
  const int b  = blockIdx.z;
  const size_t headbase = ((size_t)b * H_ + h) * CAMS_ * 64;
  const size_t vtbase   = ((size_t)b * H_ + h) * 64 * (size_t)CAMS_;

  // Q in A layout, two K-chunks of 32
  AOp qa[2];
#pragma unroll
  for (int j = 0; j < 2; ++j)
#pragma unroll
    for (int v = 0; v < 8; ++v) {
      int k = j * 32 + 16 * kb + 2 * v;
      qa[j].u[v] = *(const unsigned int*)(qr + headbase + (size_t)(c0 + m) * 64 + k);
    }

  const v8f zero8 = {0.f,0.f,0.f,0.f,0.f,0.f,0.f,0.f};
  v8f out[4];
#pragma unroll
  for (int t = 0; t < 4; ++t) out[t] = zero8;
  float rowmax[8], rowsum[8];
#pragma unroll
  for (int r = 0; r < 8; ++r) { rowmax[r] = -1e30f; rowsum[r] = 0.f; }

  const float scale = 0.125f;  // 1/sqrt(64)

  for (int j0 = 0; j0 < CAMS_; j0 += 32) {
    v8f s0 = zero8, s1 = zero8;
#pragma unroll
    for (int kc = 0; kc < 2; ++kc) {
      AOp kb0, kb1;
#pragma unroll
      for (int v = 0; v < 8; ++v) {
        int d = kc * 32 + kb * 16 + v * 2;
        kb0.u[v] = *(const unsigned int*)(kr + headbase + (size_t)(j0 + m) * 64 + d);
        kb1.u[v] = *(const unsigned int*)(kr + headbase + (size_t)(j0 + 16 + m) * 64 + d);
      }
      s0 = wmma_bf16(qa[kc], kb0, s0);
      s1 = wmma_bf16(qa[kc], kb1, s1);
    }
    // online softmax over the 32 new keys
#pragma unroll
    for (int r = 0; r < 8; ++r) {
      float a0 = s0[r] * scale, a1 = s1[r] * scale;
      float mx = fmaxf(a0, a1);
      mx = fmaxf(mx, __shfl_xor(mx, 1));
      mx = fmaxf(mx, __shfl_xor(mx, 2));
      mx = fmaxf(mx, __shfl_xor(mx, 4));
      mx = fmaxf(mx, __shfl_xor(mx, 8));
      float mnew = fmaxf(rowmax[r], mx);
      float p0 = __expf(a0 - mnew), p1 = __expf(a1 - mnew);
      float corr = __expf(rowmax[r] - mnew);
      rowmax[r] = mnew;
      float ps = p0 + p1;
      ps += __shfl_xor(ps, 1);
      ps += __shfl_xor(ps, 2);
      ps += __shfl_xor(ps, 4);
      ps += __shfl_xor(ps, 8);
      rowsum[r] = rowsum[r] * corr + ps;
#pragma unroll
      for (int t = 0; t < 4; ++t) out[t][r] *= corr;
      p_lds[r + 8 * kb][m]      = f2bf(p0);
      p_lds[r + 8 * kb][m + 16] = f2bf(p1);
    }
    __syncthreads();
    AOp pa;
#pragma unroll
    for (int v = 0; v < 8; ++v) {
      int k = 16 * kb + 2 * v;
      pa.u[v] = *(const unsigned int*)&p_lds[m][k];
    }
#pragma unroll
    for (int t = 0; t < 4; ++t) {
      AOp vb;
#pragma unroll
      for (int v = 0; v < 8; ++v) {
        int k = kb * 16 + v * 2;
        vb.u[v] = *(const unsigned int*)(vt + vtbase + (size_t)(t * 16 + m) * CAMS_ + j0 + k);
      }
      out[t] = wmma_bf16(pa, vb, out[t]);
    }
    __syncthreads();
  }

#pragma unroll
  for (int r = 0; r < 8; ++r) {
    float inv = 1.f / rowsum[r];
#pragma unroll
    for (int t = 0; t < 4; ++t) out[t][r] *= inv;
  }
#pragma unroll
  for (int t = 0; t < 4; ++t)
#pragma unroll
    for (int r = 0; r < 8; ++r)
      o_lds[r + 8 * kb][t * 16 + m] = out[t][r];
  __syncthreads();

  // apply P (forward) per camera, write merged-heads activations
  const float* Pm = VM + ((size_t)b * CAMS_ + c0 + m) * 16;
  float P[4][4];
#pragma unroll
  for (int i = 0; i < 4; ++i)
#pragma unroll
    for (int j = 0; j < 4; ++j) P[i][j] = Pm[i * 4 + j];
  size_t mbase = ((size_t)b * CAMS_ + c0 + m) * E_ + (size_t)h * 64;
#pragma unroll
  for (int g = 0; g < 8; ++g) {
    int gi = kb * 8 + g;
    float x0 = o_lds[m][4 * gi + 0];
    float x1 = o_lds[m][4 * gi + 1];
    float x2 = o_lds[m][4 * gi + 2];
    float x3 = o_lds[m][4 * gi + 3];
#pragma unroll
    for (int i = 0; i < 4; ++i) {
      float y = P[i][0] * x0 + P[i][1] * x1 + P[i][2] * x2 + P[i][3] * x3;
      merged[mbase + 4 * gi + i] = f2bf(y);
    }
  }
}

// ---------------------------------------------------------------------------
// Kernel 3: final GEMM  Y = merged @ Wo^T + bo   (fp32 out)
// grid (B*CAMS/16, E/64), block 32
// ---------------------------------------------------------------------------
__global__ __launch_bounds__(32) void out_kernel(
    const unsigned short* __restrict__ merged, const unsigned short* __restrict__ Wob,
    const float* __restrict__ bo, float* __restrict__ Y)
{
  const int lane = threadIdx.x;
  const int m  = lane & 15;
  const int kb = lane >> 4;
  const int row0 = blockIdx.x * 16;
  const int f0   = blockIdx.y * 64;

  const v8f zero8 = {0.f,0.f,0.f,0.f,0.f,0.f,0.f,0.f};
  v8f acc[4];
#pragma unroll
  for (int t = 0; t < 4; ++t) acc[t] = zero8;

  for (int kc = 0; kc < E_; kc += 32) {
    AOp A;
#pragma unroll
    for (int v = 0; v < 8; ++v) {
      int k = 16 * kb + 2 * v;
      A.u[v] = *(const unsigned int*)(merged + (size_t)(row0 + m) * E_ + kc + k);
    }
#pragma unroll
    for (int t = 0; t < 4; ++t) {
      AOp Bm;
#pragma unroll
      for (int v = 0; v < 8; ++v)
        Bm.u[v] = *(const unsigned int*)(Wob + (size_t)(f0 + t * 16 + m) * E_ + kc + kb * 16 + v * 2);
      acc[t] = wmma_bf16(A, Bm, acc[t]);
    }
  }
#pragma unroll
  for (int t = 0; t < 4; ++t) {
    int f = f0 + t * 16 + m;
    float bb = bo[f];
#pragma unroll
    for (int r = 0; r < 8; ++r)
      Y[(size_t)(row0 + r + 8 * kb) * E_ + f] = acc[t][r] + bb;
  }
}

// ---------------------------------------------------------------------------
extern "C" void kernel_launch(void* const* d_in, const int* in_sizes, int n_in,
                              void* d_out, int out_size, void* d_ws, size_t ws_size,
                              hipStream_t stream) {
  const float* vectors  = (const float*)d_in[0];
  const float* viewmats = (const float*)d_in[1];
  const float* Wq = (const float*)d_in[2]; const float* bq = (const float*)d_in[3];
  const float* Wk = (const float*)d_in[4]; const float* bk = (const float*)d_in[5];
  const float* Wv = (const float*)d_in[6]; const float* bv = (const float*)d_in[7];
  const float* Wo = (const float*)d_in[8]; const float* bo = (const float*)d_in[9];

  const size_t qkvElems = (size_t)B_ * H_ * CAMS_ * 64;  // 8M bf16 each
  const size_t xElems   = (size_t)B_ * CAMS_ * E_;       // 8M
  const size_t wElems   = (size_t)E_ * E_;               // 1M

  unsigned short* qr     = (unsigned short*)d_ws;
  unsigned short* kr     = qr + qkvElems;
  unsigned short* vt     = kr + qkvElems;
  unsigned short* merged = vt + qkvElems;
  unsigned short* Xb     = merged + qkvElems;
  unsigned short* Wqb    = Xb + xElems;
  unsigned short* Wkb    = Wqb + wElems;
  unsigned short* Wvb    = Wkb + wElems;
  unsigned short* Wob    = Wvb + wElems;

  // fp32 -> bf16 staging (elementwise, bandwidth-trivial)
  {
    int n4x = (int)(xElems / 4);
    cvt_kernel<<<dim3((n4x + 255) / 256), 256, 0, stream>>>(vectors, Xb, n4x);
    int n4w = (int)(wElems / 4);
    dim3 gw((n4w + 255) / 256);
    cvt_kernel<<<gw, 256, 0, stream>>>(Wq, Wqb, n4w);
    cvt_kernel<<<gw, 256, 0, stream>>>(Wk, Wkb, n4w);
    cvt_kernel<<<gw, 256, 0, stream>>>(Wv, Wvb, n4w);
    cvt_kernel<<<gw, 256, 0, stream>>>(Wo, Wob, n4w);
  }

  qkv_kernel<<<dim3(CAMS_ / 16, H_, B_), 32, 0, stream>>>(
      Xb, viewmats, Wqb, bq, Wkb, bk, Wvb, bv, qr, kr, vt);
  attn_kernel<<<dim3(CAMS_ / 16, H_, B_), 32, 0, stream>>>(
      qr, kr, vt, viewmats, merged);
  out_kernel<<<dim3(B_ * CAMS_ / 16, E_ / 64), 32, 0, stream>>>(
      merged, Wob, bo, (float*)d_out);
}
